// SNN_Temporal_32246614458554
// MI455X (gfx1250) — matrix operand order repaired and checked by reading, not compiled, not run on hardware
//
#include <hip/hip_runtime.h>
#include <hip/hip_bf16.h>
#include <stddef.h>

// ---------------------------------------------------------------------------
// SNN temporal network on gfx1250 using V_WMMA_F32_16X16X32_BF16.
// T=64, B=256, S=512, H=2048, A=256. BETA=0.9, THRESHOLD=1.0.
// Strategy: one-time f32->bf16 conversion of spikes + weights (all stay hot
// in 192MB L2), then per timestep three WMMA GEMM kernels with fused
// bias+LIF epilogues. Spikes {0,1} are exact in bf16; accumulation is f32.
// ---------------------------------------------------------------------------

typedef float    v8f   __attribute__((ext_vector_type(8)));
typedef __bf16   v16bf __attribute__((ext_vector_type(16)));
typedef unsigned short v16us __attribute__((ext_vector_type(16)));
typedef unsigned short v8us  __attribute__((ext_vector_type(8)));

#define TT 64
#define BB 256
#define SS 512
#define HH 2048
#define AA 256

static __device__ __forceinline__ unsigned short f2bf(float f) {
    unsigned u = __builtin_bit_cast(unsigned, f);
    unsigned r = u + 0x7FFFu + ((u >> 16) & 1u);   // round-to-nearest-even
    return (unsigned short)(r >> 16);
}

// A fragment (16x32 bf16), documented layout:
//   lane = {r=lane&15 -> M row, half=lane>>4}
//   elements j=0..7  -> K = half*8 + j        (16B chunk)
//   elements j=8..15 -> K = 16 + half*8 + j-8 (16B chunk)
static __device__ __forceinline__ v16bf load_a_bf16(const unsigned short* base, int half) {
    v8us c0 = *(const v8us*)(base + half * 8);
    v8us c1 = *(const v8us*)(base + 16 + half * 8);
    v16us w = __builtin_shufflevector(c0, c1, 0,1,2,3,4,5,6,7,8,9,10,11,12,13,14,15);
    return __builtin_bit_cast(v16bf, w);
}

// B fragment (32x16 bf16): lane holds column n=lane&15 of B = row n of W
// (W row-major [N,K]); j=0..7 -> K = half*8 + j, j=8..15 -> K = 16 + half*8 + j-8.
static __device__ __forceinline__ v16bf load_b_bf16(const unsigned short* wrow_k0, int half) {
    v8us c0 = *(const v8us*)(wrow_k0 + half * 8);
    v8us c1 = *(const v8us*)(wrow_k0 + 16 + half * 8);
    v16us raw = __builtin_shufflevector(c0, c1, 0,1,2,3,4,5,6,7,8,9,10,11,12,13,14,15);
    return __builtin_bit_cast(v16bf, raw);
}

// One fused layer: C[256,N] = A[256,K] @ W[N,K]^T, then epilogue.
//   OUT: accumulate (cur + bias) into outAcc (layer 3); else LIF + spike.
// Wave tile: 32 (M) x 64 (N); block = 8 waves; grid = (8 * N/64) / 8.
template <bool OUT>
__global__ __launch_bounds__(256)
void snn_layer(const unsigned short* __restrict__ Ain,
               const unsigned short* __restrict__ Wbf,
               const float* __restrict__ bias,
               float* __restrict__ mem,
               unsigned short* __restrict__ spkOut,
               float* __restrict__ outAcc,
               int K, int N)
{
    const int tid  = threadIdx.x;
    const int lane = tid & 31;
    const int wave = tid >> 5;
    const int w    = blockIdx.x * 8 + wave;
    const int m0   = (w & 7) * 32;          // 256/32 = 8 M-tiles
    const int n0   = (w >> 3) * 64;
    const int r    = lane & 15;
    const int half = lane >> 4;

    v8f acc[2][4];
#pragma unroll
    for (int i = 0; i < 2; ++i)
#pragma unroll
        for (int c = 0; c < 4; ++c) acc[i][c] = (v8f)0.0f;

    const unsigned short* a0p = Ain + (size_t)(m0 + r) * K;
    const unsigned short* a1p = Ain + (size_t)(m0 + 16 + r) * K;
    const unsigned short* bp[4];
#pragma unroll
    for (int c = 0; c < 4; ++c)
        bp[c] = Wbf + (size_t)(n0 + c * 16 + r) * K;

#pragma unroll 2
    for (int k0 = 0; k0 < K; k0 += 32) {
        v16bf a[2];
        a[0] = load_a_bf16(a0p + k0, half);
        a[1] = load_a_bf16(a1p + k0, half);
        v16bf b[4];
#pragma unroll
        for (int c = 0; c < 4; ++c)
            b[c] = load_b_bf16(bp[c] + k0, half);
#pragma unroll
        for (int i = 0; i < 2; ++i)
#pragma unroll
            for (int c = 0; c < 4; ++c)
                acc[i][c] = __builtin_amdgcn_wmma_f32_16x16x32_bf16(
                    false, a[i], false, b[c], (short)0, acc[i][c], false, false);
    }

    // Epilogue. C/D layout: lane -> col = n0+16c+(lane&15); vgpr rr -> row =
    // m0 + 16*i + 8*half + rr.
#pragma unroll
    for (int i = 0; i < 2; ++i) {
#pragma unroll
        for (int c = 0; c < 4; ++c) {
            const int col = n0 + c * 16 + r;
            const float bv = bias[col];
#pragma unroll
            for (int rr = 0; rr < 8; ++rr) {
                const int row = m0 + i * 16 + half * 8 + rr;
                const size_t idx = (size_t)row * N + col;
                const float cur = acc[i][c][rr] + bv;
                if (OUT) {
                    outAcc[idx] += cur;
                } else {
                    const float mo    = mem[idx];
                    const float reset = (mo > 1.0f) ? 1.0f : 0.0f;
                    const float mn    = 0.9f * mo + cur - reset;   // *THRESHOLD
                    mem[idx] = mn;
                    spkOut[idx] = (mn > 1.0f) ? (unsigned short)0x3F80
                                              : (unsigned short)0x0000;
                }
            }
        }
    }
}

// Vectorized f32 -> bf16 conversion: 8 elements per thread.
__global__ void cvt_f32_bf16_kernel(const float* __restrict__ src,
                                    unsigned short* __restrict__ dst, int n8) {
    int i = blockIdx.x * 256 + threadIdx.x;
    if (i < n8) {
        typedef float v8fv __attribute__((ext_vector_type(8)));
        v8fv v = *(const v8fv*)(src + (size_t)i * 8);
        v8us o;
#pragma unroll
        for (int j = 0; j < 8; ++j) o[j] = f2bf(v[j]);
        *(v8us*)(dst + (size_t)i * 8) = o;
    }
}

__global__ void zero_f32_kernel(float* __restrict__ p, int n) {
    int i = blockIdx.x * 256 + threadIdx.x;
    if (i < n) p[i] = 0.0f;
}

__global__ void finalize_kernel(const float* __restrict__ acc,
                                float* __restrict__ out, int n, float scale) {
    int i = blockIdx.x * 256 + threadIdx.x;
    if (i < n) out[i] = acc[i] * scale;
}

static inline size_t alignup(size_t x) { return (x + 255) & ~(size_t)255; }

extern "C" void kernel_launch(void* const* d_in, const int* in_sizes, int n_in,
                              void* d_out, int out_size, void* d_ws, size_t ws_size,
                              hipStream_t stream) {
    (void)in_sizes; (void)n_in; (void)out_size; (void)ws_size;
    const float* spikes = (const float*)d_in[0];   // [T,B,S] f32
    const float* W1     = (const float*)d_in[1];   // [H,S]
    const float* b1     = (const float*)d_in[2];   // [H]
    const float* W2     = (const float*)d_in[3];   // [H,H]
    const float* b2     = (const float*)d_in[4];   // [H]
    const float* Wo     = (const float*)d_in[5];   // [A,H]
    const float* bo     = (const float*)d_in[6];   // [A]

    // Workspace carve-out
    char* ws = (char*)d_ws;
    size_t off = 0;
    unsigned short* Xb  = (unsigned short*)(ws + off); off += alignup((size_t)TT * BB * SS * 2);
    unsigned short* W1b = (unsigned short*)(ws + off); off += alignup((size_t)HH * SS * 2);
    unsigned short* W2b = (unsigned short*)(ws + off); off += alignup((size_t)HH * HH * 2);
    unsigned short* Wob = (unsigned short*)(ws + off); off += alignup((size_t)AA * HH * 2);
    float*          mem1 = (float*)(ws + off);          off += alignup((size_t)BB * HH * 4);
    float*          mem2 = (float*)(ws + off);          off += alignup((size_t)BB * HH * 4);
    unsigned short* spk1 = (unsigned short*)(ws + off); off += alignup((size_t)BB * HH * 2);
    unsigned short* spk2 = (unsigned short*)(ws + off); off += alignup((size_t)BB * HH * 2);
    float*          oacc = (float*)(ws + off);          off += alignup((size_t)BB * AA * 4);

    // One-time f32 -> bf16 conversion (spikes streamed once; weights stay in L2)
    {
        int nx = TT * BB * SS / 8;
        int n1 = HH * SS / 8, n2 = HH * HH / 8, n3 = AA * HH / 8;
        cvt_f32_bf16_kernel<<<(nx + 255) / 256, 256, 0, stream>>>(spikes, Xb, nx);
        cvt_f32_bf16_kernel<<<(n1 + 255) / 256, 256, 0, stream>>>(W1, W1b, n1);
        cvt_f32_bf16_kernel<<<(n2 + 255) / 256, 256, 0, stream>>>(W2, W2b, n2);
        cvt_f32_bf16_kernel<<<(n3 + 255) / 256, 256, 0, stream>>>(Wo, Wob, n3);
    }
    // State init (harness does not zero workspace)
    {
        int nm = BB * HH, no = BB * AA;
        zero_f32_kernel<<<(nm + 255) / 256, 256, 0, stream>>>(mem1, nm);
        zero_f32_kernel<<<(nm + 255) / 256, 256, 0, stream>>>(mem2, nm);
        zero_f32_kernel<<<(no + 255) / 256, 256, 0, stream>>>(oacc, no);
    }

    // grid = (#M-tiles * #N-tiles) / 8 waves-per-block
    const int grid_h = (8 * (HH / 64)) / 8;   // 32 blocks, N=2048
    const int grid_a = (8 * (AA / 64)) / 8;   //  4 blocks, N=256

    for (int t = 0; t < TT; ++t) {
        const unsigned short* xt = Xb + (size_t)t * BB * SS;
        // Layer 1: K=512, N=2048, LIF -> spk1
        snn_layer<false><<<grid_h, 256, 0, stream>>>(
            xt, W1b, b1, mem1, spk1, nullptr, SS, HH);
        // Layer 2: K=2048, N=2048, LIF -> spk2
        snn_layer<false><<<grid_h, 256, 0, stream>>>(
            spk1, W2b, b2, mem2, spk2, nullptr, HH, HH);
        // Layer 3: K=2048, N=256, accumulate into oacc
        snn_layer<true ><<<grid_a, 256, 0, stream>>>(
            spk2, Wob, bo, nullptr, nullptr, oacc, HH, AA);
    }

    // out = oacc / T
    {
        int n = BB * AA;
        finalize_kernel<<<(n + 255) / 256, 256, 0, stream>>>(
            oacc, (float*)d_out, n, 1.0f / (float)TT);
    }
}
